// DeformableConv_39367670235438
// MI455X (gfx1250) — matrix operand order repaired and checked by reading, not compiled, not run on hardware
//
#include <hip/hip_runtime.h>
#include <hip/hip_bf16.h>

typedef _Float16 h16;
typedef __attribute__((ext_vector_type(16))) _Float16 v16h;
typedef __attribute__((ext_vector_type(8)))  _Float16 v8h;
typedef __attribute__((ext_vector_type(8)))  float    v8f;

#define B_   4
#define C_   64
#define H_   160
#define W_   160
#define HW_  (H_ * W_)        // 25600
#define DG_  4
#define KK_  9
#define CG_  16
#define NT_  (HW_ / 16)       // 1600 pixel tiles per batch (W divisible by 16)

// ---- workspace layout (bytes, all 16B aligned) ----
#define WS_WA    0u                         // 64*64 h16        = 8192
#define WS_BA    (WS_WA   + 8192u)          // 64 f32           = 256
#define WS_WB    (WS_BA   + 256u)           // 112*576 h16      = 129024
#define WS_BCOM  (WS_WB   + 129024u)        // 112 f32 -> pad   = 512
#define WS_WD    (WS_BCOM + 512u)           // 64*576 h16       = 73728
#define WS_FEAT  (WS_WD   + 73728u)         // B*HW*64 h16      = 13107200
#define WS_OFFT  (WS_FEAT + 13107200u)      // B*HW*64 h16      = 13107200
#define WS_DY    (WS_OFFT + 13107200u)      // B*36*HW f32      = 14745600
#define WS_DX    (WS_DY   + 14745600u)
#define WS_MASK  (WS_DX   + 14745600u)      // total ~70.7 MB

// ---------- WMMA fragment loaders (contiguous-K order, same for A and B) ----------
// A 16x32 f16: lane holds row m=lane&15; halves = 16 contiguous K starting at hi*16
__device__ __forceinline__ v16h ldA16(const h16* row32, int hi8) {
  union { v16h v; v8h h[2]; } u;
  u.h[0] = *(const v8h*)(row32 + 2 * hi8);
  u.h[1] = *(const v8h*)(row32 + 2 * hi8 + 8);
  return u.v;
}
// B 32x16 f16: lane holds col n=lane&15; halves = 16 contiguous K starting at hi*16
__device__ __forceinline__ v16h ldB16c(const h16* p16) {
  union { v16h v; v8h h[2]; } u;
  u.h[0] = *(const v8h*)(p16);
  u.h[1] = *(const v8h*)(p16 + 8);
  return u.v;
}

// ---------- K0: fold BN, reorder & convert weights to f16 ----------
__global__ void k_prep(const float* __restrict__ w_off, const float* __restrict__ gamma,
                       const float* __restrict__ beta,  const float* __restrict__ mean,
                       const float* __restrict__ var,   const float* __restrict__ w_com,
                       const float* __restrict__ b_com, const float* __restrict__ w_dcn,
                       h16* wA, float* bA, h16* wB, float* bcom, h16* wD) {
  int idx = blockIdx.x * 256 + threadIdx.x;
  if (idx < 64 * 64) {
    int o = idx >> 6, i = idx & 63;
    float inv = gamma[o] * rsqrtf(var[o] + 1e-5f);
    wA[idx] = (h16)(w_off[idx] * inv);
    if (i == 0) bA[o] = beta[o] - mean[o] * inv;
  }
  int j = idx - 64 * 64;                       // conv weights: [o][tap*64+i] <- [o][i][tap]
  if (j >= 0 && j < 112 * 576) {
    int o = j / 576, r = j % 576;
    int tap = r >> 6, i = r & 63;
    wB[j] = (h16)((o < 108) ? w_com[o * 576 + i * 9 + tap] : 0.f);
    if (r == 0) bcom[o] = (o < 108) ? b_com[o] : 0.f;
  }
  int j2 = idx - 64 * 64 - 112 * 576;          // w_dcn is already [o][i*9+tap]
  if (j2 >= 0 && j2 < 64 * 576) wD[j2] = (h16)w_dcn[j2];
}

// ---------- K1: NCHW f32 -> pixel-major f16 transpose via LDS ----------
__global__ __launch_bounds__(256) void k_transpose(const float* __restrict__ fea, h16* __restrict__ feaT) {
  __shared__ float lds[64][65];
  int b  = blockIdx.x / 400;
  int p0 = (blockIdx.x % 400) * 64;
  const float* src = fea + (size_t)b * C_ * HW_ + p0;
#pragma unroll
  for (int r = 0; r < 16; ++r) {
    int idx = r * 256 + threadIdx.x;
    lds[idx >> 6][idx & 63] = src[(size_t)(idx >> 6) * HW_ + (idx & 63)];
  }
  __syncthreads();
  h16* dst = feaT + ((size_t)b * HW_ + p0) * 64;
#pragma unroll
  for (int r = 0; r < 16; ++r) {
    int idx = r * 256 + threadIdx.x;
    dst[(size_t)(idx >> 6) * 64 + (idx & 63)] = (h16)lds[idx & 63][idx >> 6];
  }
}

// ---------- K2: off_feat = (BN-folded w_off) x fea, f16 out, pixel-major ----------
__global__ __launch_bounds__(256) void k_off_gemm(const h16* __restrict__ feaT, const h16* __restrict__ wA,
                                                  const float* __restrict__ bA, h16* __restrict__ offT) {
  int lane = threadIdx.x & 31, wv = threadIdx.x >> 5;
  int gw = blockIdx.x * 8 + wv;                 // B_*NT_*4 waves total
  int b  = gw / (NT_ * 4);
  int r  = gw % (NT_ * 4);
  int nt = r >> 2, mt = r & 3;
  int n = lane & 15, hi = lane >> 4, hi8 = hi * 8;
  int p = nt * 16 + n;
  const h16* arow = wA + (mt * 16 + n) * 64;
  const h16* brow = feaT + ((size_t)b * HW_ + p) * 64 + hi * 16;
  v8f acc = {};
#pragma unroll
  for (int ks = 0; ks < 2; ++ks) {
    v16h a  = ldA16(arow + ks * 32, hi8);
    v16h bb = ldB16c(brow + ks * 32);
    acc = __builtin_amdgcn_wmma_f32_16x16x32_f16(false, a, false, bb, (short)0, acc, false, false);
  }
  union { v8f v; float f[8]; } res; res.v = acc;
  h16* orow = offT + ((size_t)b * HW_ + p) * 64;
#pragma unroll
  for (int v = 0; v < 8; ++v) {
    int o = mt * 16 + hi8 + v;
    orow[o] = (h16)(res.f[v] + bA[o]);
  }
}

// ---------- K3: implicit-im2col 3x3 conv 64->108, fused bias/sigmoid/scatter ----------
__global__ __launch_bounds__(256) void k_com_conv(const h16* __restrict__ offT, const h16* __restrict__ wB,
                                                  const float* __restrict__ bcom, float* __restrict__ dy,
                                                  float* __restrict__ dx, float* __restrict__ mk) {
  int lane = threadIdx.x & 31, wv = threadIdx.x >> 5;
  int gw = blockIdx.x * 8 + wv;                 // B_*NT_*7 waves total
  int b  = gw / (NT_ * 7);
  int r  = gw % (NT_ * 7);
  int nt = r / 7, mt = r % 7;
  int n = lane & 15, hi = lane >> 4, hi8 = hi * 8;
  int p0 = nt * 16, h = p0 / W_, w0 = p0 % W_;
  const h16* arow  = wB + (mt * 16 + n) * 576;
  const h16* bbase = offT + (size_t)b * HW_ * 64;
  v8f acc = {};
#pragma unroll
  for (int tap = 0; tap < 9; ++tap) {
    int ty = tap / 3 - 1, tx = tap % 3 - 1;
    int hh = h + ty, ww = w0 + n + tx;
    bool ok = (hh >= 0) & (hh < H_) & (ww >= 0) & (ww < W_);
#pragma unroll
    for (int ks = 0; ks < 2; ++ks) {
      v16h a  = ldA16(arow + tap * 64 + ks * 32, hi8);
      v16h bb = {};
      if (ok) bb = ldB16c(bbase + (size_t)(hh * W_ + ww) * 64 + hi * 16 + ks * 32);
      acc = __builtin_amdgcn_wmma_f32_16x16x32_f16(false, a, false, bb, (short)0, acc, false, false);
    }
  }
  union { v8f v; float f[8]; } res; res.v = acc;
  int p = p0 + n;
  size_t gb = (size_t)b * DG_ * KK_ * HW_;
#pragma unroll
  for (int v = 0; v < 8; ++v) {
    int ch = mt * 16 + hi8 + v;
    if (ch >= 108) continue;
    float x = res.f[v] + bcom[ch];
    if (ch < 72) {                                 // interleaved dy/dx: ch = g*18 + k*2 + comp
      int g = ch / 18, rr = ch % 18, k = rr >> 1;
      size_t o = gb + ((size_t)g * KK_ + k) * HW_ + p;
      if ((rr & 1) == 0) dy[o] = x; else dx[o] = x;
    } else {                                       // mask: sigmoid
      int t = ch - 72, g = t / 9, k = t % 9;
      mk[gb + ((size_t)g * KK_ + k) * HW_ + p] = 1.f / (1.f + __expf(-x));
    }
  }
}

// ---------- K4: bilinear gather -> LDS panel -> K=576 GEMM, fused ReLU+residual ----------
__global__ __launch_bounds__(128) void k_dcn(const float* __restrict__ fea, const float* __restrict__ dy,
                                             const float* __restrict__ dx, const float* __restrict__ mk,
                                             const h16* __restrict__ wD, const float* __restrict__ b_dcn,
                                             float* __restrict__ out) {
  __shared__ __align__(16) h16 vals[16 * 584];   // 16 px rows, 576 K + 8 pad (bank-conflict free)
  int b  = blockIdx.x / NT_;
  int nt = blockIdx.x % NT_;
  int p0 = nt * 16, h = p0 / W_, w0 = p0 % W_;

  for (int t = threadIdx.x; t < 16 * DG_ * KK_; t += 128) {
    int px = t / 36, gk = t % 36, g = gk / 9, k = gk % 9;
    int p = p0 + px;
    size_t off = ((size_t)(b * DG_ + g) * KK_ + k) * HW_ + p;
    float pyv = dy[off] + (float)(k / 3 - 1) + (float)h;
    float pxv = dx[off] + (float)(k % 3 - 1) + (float)(w0 + px);
    float mv  = mk[off];
    float y0f = floorf(pyv), x0f = floorf(pxv);
    float wy = pyv - y0f, wx = pxv - x0f;
    int y0 = (int)y0f, x0 = (int)x0f, y1 = y0 + 1, x1 = x0 + 1;
    float vy0 = (y0 >= 0 && y0 < H_) ? 1.f : 0.f;
    float vy1 = (y1 >= 0 && y1 < H_) ? 1.f : 0.f;
    float vx0 = (x0 >= 0 && x0 < W_) ? 1.f : 0.f;
    float vx1 = (x1 >= 0 && x1 < W_) ? 1.f : 0.f;
    int cy0 = min(max(y0, 0), H_ - 1), cy1 = min(max(y1, 0), H_ - 1);
    int cx0 = min(max(x0, 0), W_ - 1), cx1 = min(max(x1, 0), W_ - 1);
    float w00 = (1.f - wy) * (1.f - wx) * vy0 * vx0 * mv;
    float w01 = (1.f - wy) * wx * vy0 * vx1 * mv;
    float w10 = wy * (1.f - wx) * vy1 * vx0 * mv;
    float w11 = wy * wx * vy1 * vx1 * mv;
    int i00 = cy0 * W_ + cx0, i01 = cy0 * W_ + cx1;
    int i10 = cy1 * W_ + cx0, i11 = cy1 * W_ + cx1;
    const float* fg = fea + ((size_t)b * C_ + g * CG_) * HW_;
    h16* dst = vals + px * 584 + g * CG_ * 9 + k;   // K ordering: (g*16+c)*9 + k
#pragma unroll
    for (int c = 0; c < CG_; ++c) {
      const float* fc = fg + (size_t)c * HW_;
      float v = w00 * fc[i00] + w01 * fc[i01] + w10 * fc[i10] + w11 * fc[i11];
      dst[c * 9] = (h16)v;
    }
  }
  __syncthreads();

  int lane = threadIdx.x & 31, wv = threadIdx.x >> 5;  // 4 waves = 4 M tiles
  int n = lane & 15, hi = lane >> 4, hi8 = hi * 8;
  const h16* arow = wD + (wv * 16 + n) * 576;
  const h16* brow = vals + n * 584 + hi * 16;
  v8f acc = {};
#pragma unroll
  for (int ks = 0; ks < 18; ++ks) {                    // K = 576 = 18 x 32
    v16h a  = ldA16(arow + ks * 32, hi8);
    v16h bb = ldB16c(brow + ks * 32);
    acc = __builtin_amdgcn_wmma_f32_16x16x32_f16(false, a, false, bb, (short)0, acc, false, false);
  }
  union { v8f v; float f[8]; } res; res.v = acc;
  int p = p0 + n;
#pragma unroll
  for (int v = 0; v < 8; ++v) {
    int o = wv * 16 + hi8 + v;
    float rr = fmaxf(res.f[v] + b_dcn[o], 0.f);
    size_t oo = ((size_t)b * C_ + o) * HW_ + p;
    out[oo] = fea[oo] + rr;
  }
}

extern "C" void kernel_launch(void* const* d_in, const int* in_sizes, int n_in,
                              void* d_out, int out_size, void* d_ws, size_t ws_size,
                              hipStream_t stream) {
  const float* fea   = (const float*)d_in[0];
  const float* w_off = (const float*)d_in[1];
  const float* gamma = (const float*)d_in[2];
  const float* beta  = (const float*)d_in[3];
  const float* mean  = (const float*)d_in[4];
  const float* var   = (const float*)d_in[5];
  const float* w_com = (const float*)d_in[6];
  const float* b_com = (const float*)d_in[7];
  const float* w_dcn = (const float*)d_in[8];
  const float* b_dcn = (const float*)d_in[9];
  float* out = (float*)d_out;
  char*  ws  = (char*)d_ws;

  h16*   wA   = (h16*)  (ws + WS_WA);
  float* bA   = (float*)(ws + WS_BA);
  h16*   wB   = (h16*)  (ws + WS_WB);
  float* bcom = (float*)(ws + WS_BCOM);
  h16*   wD   = (h16*)  (ws + WS_WD);
  h16*   feaT = (h16*)  (ws + WS_FEAT);
  h16*   offT = (h16*)  (ws + WS_OFFT);
  float* dyp  = (float*)(ws + WS_DY);
  float* dxp  = (float*)(ws + WS_DX);
  float* mkp  = (float*)(ws + WS_MASK);

  k_prep<<<412, 256, 0, stream>>>(w_off, gamma, beta, mean, var, w_com, b_com, w_dcn,
                                  wA, bA, wB, bcom, wD);
  k_transpose<<<B_ * 400, 256, 0, stream>>>(fea, feaT);
  k_off_gemm<<<(B_ * NT_ * 4) / 8, 256, 0, stream>>>(feaT, wA, bA, offT);
  k_com_conv<<<(B_ * NT_ * 7) / 8, 256, 0, stream>>>(offT, wB, bcom, dyp, dxp, mkp);
  k_dcn<<<B_ * NT_, 128, 0, stream>>>(fea, dyp, dxp, mkp, wD, b_dcn, out);
}
